// GlobalAttention_76235669504398
// MI455X (gfx1250) — matrix operand order repaired and checked by reading, not compiled, not run on hardware
//
#include <hip/hip_runtime.h>
#include <hip/hip_bf16.h>

#define B_ 32
#define S_ 4096
#define H_ 512
#define ROWPAD 520   // shorts per LDS row (512 + 8 pad -> conflict-free b128 reads)

typedef __attribute__((ext_vector_type(16))) __bf16          v16bf;
typedef __attribute__((ext_vector_type(8)))  float           v8f;
typedef __attribute__((ext_vector_type(8)))  unsigned short  us8;
typedef __attribute__((ext_vector_type(4)))  unsigned short  us4;
typedef __attribute__((ext_vector_type(4)))  float           f4;

union Frag16 { us8 h[2]; v16bf v; };

// Hardware V_TANH_F32 (gfx1250 TRANS op) if the builtin is exposed; else libm.
#if __has_builtin(__builtin_amdgcn_tanhf)
  #define FAST_TANH(x) __builtin_amdgcn_tanhf(x)
#elif __has_builtin(__builtin_amdgcn_tanh_f32)
  #define FAST_TANH(x) __builtin_amdgcn_tanh_f32(x)
#else
  #define FAST_TANH(x) tanhf(x)
#endif

__device__ __forceinline__ unsigned short f2bf(float f) {
    unsigned int u = __builtin_bit_cast(unsigned int, f);
    unsigned int r = u + 0x7FFFu + ((u >> 16) & 1u);   // round-to-nearest-even
    return (unsigned short)(r >> 16);
}

// ---------------- Kernel 1: h_proj[b,o] = hidden[b,:] . W_attn[o,0:H] + b_attn[o]
__global__ __launch_bounds__(256) void ga_hproj(const float* __restrict__ hidden,
                                                const float* __restrict__ W,
                                                const float* __restrict__ bias,
                                                float* __restrict__ hproj) {
    int idx = blockIdx.x * 256 + threadIdx.x;     // B*H
    int b = idx >> 9, o = idx & (H_ - 1);
    const float* hrow = hidden + b * H_;
    const float* wrow = W + (size_t)o * 2 * H_;
    float acc = bias[o];
    #pragma unroll 8
    for (int h = 0; h < H_; ++h) acc = fmaf(hrow[h], wrow[h], acc);
    hproj[idx] = acc;
}

// ---------------- Kernel 2: We_bf16[n,k] = bf16(W_attn[n, H+k])
__global__ __launch_bounds__(256) void ga_cvt(const float* __restrict__ W,
                                              unsigned short* __restrict__ WeBf) {
    int idx = blockIdx.x * 256 + threadIdx.x;     // H*H
    int n = idx >> 9, k = idx & (H_ - 1);
    WeBf[idx] = f2bf(W[(size_t)n * 2 * H_ + H_ + k]);
}

// ---------------- Kernel 3: scores[b,s] = v . tanh(enc[b,s,:] @ We^T + h_proj[b,:])
// One wave (32 lanes) per 16-row s-tile; 2 waves per block.
__global__ __launch_bounds__(64) void ga_scores(const float* __restrict__ enc,
                                                const float* __restrict__ hproj,
                                                const float* __restrict__ vvec,
                                                const unsigned short* __restrict__ WeBf,
                                                float* __restrict__ scores) {
    __shared__ unsigned short ldsA[2][16][ROWPAD];
    const int lane = threadIdx.x & 31;
    const int wave = threadIdx.x >> 5;
    const int tile = blockIdx.x * 2 + wave;           // s-tile id in [0, B*S/16)
    const int b  = tile >> 8;                          // S/16 = 256 tiles per batch
    const int s0 = (tile & 255) << 4;

    // Stage 16 rows x 512 f32 -> bf16 into LDS (wave-private region, coalesced 16B chunks)
    const float* src = enc + ((size_t)(b * S_ + s0)) * H_;
    #pragma unroll 4
    for (int it = 0; it < 64; ++it) {
        int idx4 = it * 32 + lane;                     // float4 units, 128 per row
        int row = idx4 >> 7, c4 = idx4 & 127;
        f4 d = *(const f4*)(src + (size_t)row * H_ + c4 * 4);
        us4 o; o.x = f2bf(d.x); o.y = f2bf(d.y); o.z = f2bf(d.z); o.w = f2bf(d.w);
        *(us4*)&ldsA[wave][row][c4 * 4] = o;
    }

    const int hi   = (lane >= 16) ? 1 : 0;
    const int arow = lane & 15;                        // A row / C column within tile
    float sacc[8] = {0.f, 0.f, 0.f, 0.f, 0.f, 0.f, 0.f, 0.f};

    for (int nt = 0; nt < 32; ++nt) {
        v8f acc = {0.f, 0.f, 0.f, 0.f, 0.f, 0.f, 0.f, 0.f};
        const int ncol = nt * 16 + arow;               // this lane's output column N
        const unsigned short* brow = WeBf + (size_t)ncol * H_;
        #pragma unroll 4
        for (int kt = 0; kt < 16; ++kt) {
            // A fragment (ISA 16-bit A 16x32): two contiguous K-chunks per lane
            Frag16 a;
            int c0 = kt * 32 + hi * 8;
            a.h[0] = *(const us8*)&ldsA[wave][arow][c0];
            a.h[1] = *(const us8*)&ldsA[wave][arow][c0 + 16];
            // B fragment: two 8-K chunks at row (nt*16 + lane%16), offsets hi*8 and hi*8+16
            Frag16 bb;
            int kb = kt * 32 + hi * 8;
            bb.h[0] = *(const us8*)(brow + kb);
            bb.h[1] = *(const us8*)(brow + kb + 16);
            acc = __builtin_amdgcn_wmma_f32_16x16x32_bf16(
                false, a.v, false, bb.v, (short)0, acc, false, false);
        }
        // Fused epilogue: energy = tanh(e + h_proj), partial score += energy * v
        float hp = hproj[b * H_ + ncol];
        float vv = vvec[ncol];
        #pragma unroll
        for (int r = 0; r < 8; ++r) {                  // C: VGPR r -> M = r + 8*hi, N = ncol
            float e = acc[r] + hp;
            sacc[r] = fmaf(FAST_TANH(e), vv, sacc[r]);
        }
    }

    // Butterfly-reduce over the 16 columns held by each lane half
    #pragma unroll
    for (int r = 0; r < 8; ++r) {
        float x = sacc[r];
        for (int m = 1; m < 16; m <<= 1) x += __shfl_xor(x, m, 32);
        sacc[r] = x;
    }
    if (lane == 0) {
        #pragma unroll
        for (int r = 0; r < 8; ++r) scores[b * S_ + s0 + r] = sacc[r];
    }
    if (lane == 16) {
        #pragma unroll
        for (int r = 0; r < 8; ++r) scores[b * S_ + s0 + 8 + r] = sacc[r];
    }
}

// ---------------- Kernel 4: softmax over S per batch -> attention weights (d_out tail)
__global__ __launch_bounds__(256) void ga_softmax(const float* __restrict__ scores,
                                                  float* __restrict__ attn) {
    __shared__ float red[256];
    const int b = blockIdx.x, t = threadIdx.x;
    const float* sc = scores + b * S_;
    float mx = -3.0e38f;
    for (int s = t; s < S_; s += 256) mx = fmaxf(mx, sc[s]);
    red[t] = mx; __syncthreads();
    for (int st = 128; st > 0; st >>= 1) { if (t < st) red[t] = fmaxf(red[t], red[t + st]); __syncthreads(); }
    mx = red[0]; __syncthreads();
    float sum = 0.f;
    for (int s = t; s < S_; s += 256) sum += __expf(sc[s] - mx);
    red[t] = sum; __syncthreads();
    for (int st = 128; st > 0; st >>= 1) { if (t < st) red[t] += red[t + st]; __syncthreads(); }
    float inv = 1.0f / red[0];
    for (int s = t; s < S_; s += 256) attn[b * S_ + s] = __expf(sc[s] - mx) * inv;
}

// ---------------- Kernel 5: context[b,h] = sum_s attn[b,s] * enc[b,s,h]
__global__ __launch_bounds__(256) void ga_context(const float* __restrict__ enc,
                                                  const float* __restrict__ attn,
                                                  float* __restrict__ ctx) {
    __shared__ float red[256];
    const int b = blockIdx.x >> 2, hb = blockIdx.x & 3, t = threadIdx.x;
    const int hx = t & 127, sy = t >> 7;               // 2-way s-split per h
    const int h = hb * 128 + hx;
    const float* w = attn + b * S_;
    const float* e = enc + ((size_t)b * S_) * H_ + h;
    float acc = 0.f;
    for (int s = sy; s < S_; s += 2) acc = fmaf(w[s], e[(size_t)s * H_], acc);
    red[t] = acc; __syncthreads();
    if (sy == 0) ctx[b * H_ + h] = red[t] + red[t + 128];
}

extern "C" void kernel_launch(void* const* d_in, const int* in_sizes, int n_in,
                              void* d_out, int out_size, void* d_ws, size_t ws_size,
                              hipStream_t stream) {
    const float* hidden = (const float*)d_in[0];       // (B,H)
    const float* enc    = (const float*)d_in[1];       // (B,S,H)
    const float* W      = (const float*)d_in[2];       // (H,2H)
    const float* bias   = (const float*)d_in[3];       // (H,)
    const float* v      = (const float*)d_in[4];       // (H,)

    float* ctx  = (float*)d_out;                       // context: B*H
    float* attn = (float*)d_out + B_ * H_;             // weights: B*S

    float* hproj            = (float*)d_ws;                    // B*H f32
    float* scores           = hproj + B_ * H_;                 // B*S f32
    unsigned short* WeBf    = (unsigned short*)(scores + B_ * S_); // H*H bf16

    ga_hproj  <<<(B_ * H_) / 256, 256, 0, stream>>>(hidden, W, bias, hproj);
    ga_cvt    <<<(H_ * H_) / 256, 256, 0, stream>>>(W, WeBf);
    ga_scores <<<(B_ * S_) / 32, 64, 0, stream>>>(enc, hproj, v, WeBf, scores);
    ga_softmax<<<B_, 256, 0, stream>>>(scores, attn);
    ga_context<<<B_ * 4, 256, 0, stream>>>(enc, attn, ctx);
}
